// DeepTypedGraphNet_30416958390479
// MI455X (gfx1250) — matrix operand order repaired and checked by reading, not compiled, not run on hardware
//
#include <hip/hip_runtime.h>

// ---------------------------------------------------------------------------
// DeepTypedGraphNet (GraphCast-style GNN) on MI455X (gfx1250, wave32, WMMA).
//
// GEMMs: v_wmma_f32_16x16x32_f16, 8 waves/block, block tile 256x64,
// wave tile 32x64 (8 f32 accumulators). B (weights) is pre-packed into the
// ISA B-fragment layout and staged through double-buffered LDS so all 8
// waves share one global fetch per k-step (WGP has 320KB LDS; we use 8KB).
// Activations stream as f16; LayerNorm / residuals / segment-sum in f32.
// Segment-sum is fused into the edge LayerNorm kernel via global f32 atomics
// (agg is 16MB -> L2-resident on the 192MB L2).
// ---------------------------------------------------------------------------

typedef _Float16 half8  __attribute__((ext_vector_type(8)));
typedef _Float16 half16 __attribute__((ext_vector_type(16)));
typedef float    float8 __attribute__((ext_vector_type(8)));

#define GN_N   16384
#define GN_E   131072
#define GN_L   256
#define GN_S   4

// ---------------------------------------------------------------------------
// Pack a row-major f32 weight W[K,N] into f16 WMMA B-fragments.
// Fragment (nt, kt): 512 halfs; lane l < 16 holds N = nt*16+l, K = kt*32+(h<8 ? h : h+8)
//                    lane l >= 16 holds N = nt*16+l-16, K = kt*32+8+(h<8 ? h : h+8)
// idx = ((nt*kTiles + kt)*32 + lane)*16 + h  (each lane's 16 halfs contiguous)
// k >= K (padding rows) are zero-filled.
// ---------------------------------------------------------------------------
__global__ __launch_bounds__(256)
void pack_weight_kernel(const float* __restrict__ W, _Float16* __restrict__ dst,
                        int K, int Kp, int N) {
  const int kTiles = Kp >> 5;
  const int total = (N >> 4) * kTiles * 512;
  int idx = blockIdx.x * 256 + threadIdx.x;
  if (idx >= total) return;
  const int h    = idx & 15;
  const int lane = (idx >> 4) & 31;
  const int rest = idx >> 9;
  const int kt   = rest % kTiles;
  const int nt   = rest / kTiles;
  const int n = nt * 16 + (lane & 15);
  const int k = kt * 32 + ((lane < 16) ? 0 : 8) + ((h < 8) ? h : h + 8);
  dst[idx] = (k < K) ? (_Float16)W[(size_t)k * N + n] : (_Float16)0.f;
}

// Convert raw features [M,K] f32 -> [M,32] f16 zero-padded (K in {3,4}).
__global__ __launch_bounds__(256)
void convert_pad_kernel(const float* __restrict__ src, _Float16* __restrict__ dst,
                        int K, int M) {
  int idx = blockIdx.x * 256 + threadIdx.x;
  if (idx >= M * 32) return;
  const int row = idx >> 5;
  const int k   = idx & 31;
  dst[idx] = (k < K) ? (_Float16)src[row * K + k] : (_Float16)0.f;
}

__global__ __launch_bounds__(256)
void zero_f4_kernel(float4* __restrict__ p, int n4) {
  int i = blockIdx.x * 256 + threadIdx.x;
  if (i < n4) p[i] = make_float4(0.f, 0.f, 0.f, 0.f);
}

// ---------------------------------------------------------------------------
// WMMA GEMM: C[M,N] = epi(A[M,Kp] @ W + bias)
//   A: f16 row-major, stride Kp (mult of 32). M % 256 == 0, N % 64 == 0.
//   Block = 8 waves, tile 256x64; wave tile 32x64 (2 M-subtiles x 4 N-subtiles).
//   B fragments double-buffered through LDS (4KB/k-step), one barrier/k-step.
//   EPI 0: ReLU, store f16.   EPI 1: no act, store f32.
// ---------------------------------------------------------------------------
union AFrag { half16 v; half8 h[2]; };

template <int EPI>
__global__ __launch_bounds__(256)
void gemm_wmma_kernel(const _Float16* __restrict__ A,
                      const _Float16* __restrict__ Bp,
                      const float* __restrict__ bias,
                      void* __restrict__ Cout,
                      int M, int Kp, int N) {
  __shared__ __attribute__((aligned(32))) _Float16 sB[2][2048];

  const int lane  = threadIdx.x & 31;
  const int wave  = threadIdx.x >> 5;
  const int mBase = blockIdx.x * 256 + wave * 32;
  const int nBase = blockIdx.y * 64;
  const int row0  = mBase + (lane & 15);
  const int kOff  = (lane < 16) ? 0 : 8;        // A-fragment lane K phase
  const int kTiles = Kp >> 5;
  const size_t nFragStride = (size_t)kTiles * 512;

  // B staging assignment for this thread: fragment snt, 8-half chunk sch.
  const int snt = threadIdx.x >> 6;
  const int sch = threadIdx.x & 63;
  const _Float16* __restrict__ Bstage =
      Bp + (size_t)((nBase >> 4) + snt) * nFragStride + (size_t)sch * 8;

  float8 acc[2][4] = {};

  const _Float16* __restrict__ Arow0 = A + (size_t)row0 * Kp;
  const _Float16* __restrict__ Arow1 = Arow0 + (size_t)16 * Kp;

  // Prologue: stage k-tile 0 into buffer 0.
  *(half8*)(&sB[0][snt * 512 + sch * 8]) = *(const half8*)(Bstage);
  __syncthreads();

  for (int kt = 0; kt < kTiles; ++kt) {
    const int cur = kt & 1;
    if (kt + 1 < kTiles) {
      *(half8*)(&sB[cur ^ 1][snt * 512 + sch * 8]) =
          *(const half8*)(Bstage + (size_t)(kt + 1) * 512);
    }

    AFrag a0, a1;
    const int k0 = kt * 32 + kOff;
    a0.h[0] = *(const half8*)(Arow0 + k0);        // K = k0 .. k0+7
    a0.h[1] = *(const half8*)(Arow0 + k0 + 16);   // K = k0+16 .. k0+23
    a1.h[0] = *(const half8*)(Arow1 + k0);
    a1.h[1] = *(const half8*)(Arow1 + k0 + 16);

#pragma unroll
    for (int nt = 0; nt < 4; ++nt) {
      const half16 b = *(const half16*)(&sB[cur][(nt * 32 + lane) * 16]);
      acc[0][nt] = __builtin_amdgcn_wmma_f32_16x16x32_f16(
          false, a0.v, false, b, (short)0, acc[0][nt], false, false);
      acc[1][nt] = __builtin_amdgcn_wmma_f32_16x16x32_f16(
          false, a1.v, false, b, (short)0, acc[1][nt], false, false);
    }
    __syncthreads();
  }

  // C/D layout: VGPR r -> M = r (lanes 0-15) or 8+r (lanes 16-31), N = lane%16
  const int colLane = lane & 15;
  const int rowSel  = (lane < 16) ? 0 : 8;
#pragma unroll
  for (int am = 0; am < 2; ++am) {
    const int orowBase = mBase + am * 16 + rowSel;
#pragma unroll
    for (int nt = 0; nt < 4; ++nt) {
      const int ocol = nBase + nt * 16 + colLane;
      const float bv = bias[ocol];
#pragma unroll
      for (int r = 0; r < 8; ++r) {
        const int orow = orowBase + r;
        float v = acc[am][nt][r] + bv;
        if (EPI == 0) {
          v = v > 0.f ? v : 0.f;
          ((_Float16*)Cout)[(size_t)orow * N + ocol] = (_Float16)v;
        } else {
          ((float*)Cout)[(size_t)orow * N + ocol] = v;
        }
      }
    }
  }
}

// ---------------------------------------------------------------------------
// LayerNorm over 256 cols, one wave per row (32 lanes x 8 floats).
//   nv = (y - mu) * rsqrt(var + eps) * ls + lb
//   state = addResidual ? state + nv : nv   (f32 + f16 mirror)
//   if agg != null: atomicAdd(agg[recv[row]*256 + c], nv)   (segment_sum)
// ---------------------------------------------------------------------------
__global__ __launch_bounds__(256)
void layernorm_kernel(const float* __restrict__ y,
                      const float* __restrict__ ls, const float* __restrict__ lb,
                      float* __restrict__ state_f32, _Float16* __restrict__ state_f16,
                      float* __restrict__ agg, const int* __restrict__ recv,
                      int M, int addResidual) {
  const int row  = blockIdx.x * 8 + (threadIdx.x >> 5);
  const int lane = threadIdx.x & 31;
  if (row >= M) return;
  const float* yr = y + (size_t)row * 256;
  float v[8];
  float s = 0.f, sq = 0.f;
#pragma unroll
  for (int i = 0; i < 8; ++i) {
    v[i] = yr[lane * 8 + i];
    s += v[i];
    sq += v[i] * v[i];
  }
#pragma unroll
  for (int off = 16; off >= 1; off >>= 1) {
    s  += __shfl_xor(s,  off, 32);
    sq += __shfl_xor(sq, off, 32);
  }
  const float mu   = s * (1.f / 256.f);
  const float var  = sq * (1.f / 256.f) - mu * mu;
  const float rstd = rsqrtf(var + 1e-6f);

  const int r = (agg != nullptr) ? recv[row] : 0;
#pragma unroll
  for (int i = 0; i < 8; ++i) {
    const int c = lane * 8 + i;
    const float nv = (v[i] - mu) * rstd * ls[c] + lb[c];
    const size_t si = (size_t)row * 256 + c;
    const float st = addResidual ? (state_f32[si] + nv) : nv;
    state_f32[si] = st;
    state_f16[si] = (_Float16)st;
    if (agg != nullptr) atomicAdd(&agg[(size_t)r * 256 + c], nv);
  }
}

// ein[e] = concat(edges_f16[e], nodes_f16[senders[e]], nodes_f16[receivers[e]])
// 96 x 16B chunks per edge.
__global__ __launch_bounds__(256)
void gather_concat_kernel(const _Float16* __restrict__ edges,
                          const _Float16* __restrict__ nodes,
                          const int* __restrict__ senders,
                          const int* __restrict__ receivers,
                          _Float16* __restrict__ ein, int E) {
  int t = blockIdx.x * 256 + threadIdx.x;
  const int total = E * 96;
  if (t >= total) return;
  const int e = t / 96;
  const int c = t % 96;
  const uint4* src;
  if (c < 32)       src = (const uint4*)(edges + (size_t)e * 256) + c;
  else if (c < 64)  src = (const uint4*)(nodes + (size_t)senders[e]   * 256) + (c - 32);
  else              src = (const uint4*)(nodes + (size_t)receivers[e] * 256) + (c - 64);
  ((uint4*)(ein + (size_t)e * 768))[c] = *src;
}

// nin[n] = concat(nodes_f16[n], f16(agg[n]))  — 64 x 16B chunks per node.
__global__ __launch_bounds__(256)
void node_concat_kernel(const _Float16* __restrict__ nodes,
                        const float* __restrict__ agg,
                        _Float16* __restrict__ nin, int N) {
  int t = blockIdx.x * 256 + threadIdx.x;
  if (t >= N * 64) return;
  const int n = t >> 6;
  const int c = t & 63;
  if (c < 32) {
    ((uint4*)(nin + (size_t)n * 512))[c] = ((const uint4*)(nodes + (size_t)n * 256))[c];
  } else {
    const float* a = agg + (size_t)n * 256 + (c - 32) * 8;
    half8 h;
#pragma unroll
    for (int i = 0; i < 8; ++i) h[i] = (_Float16)a[i];
    *(half8*)(nin + (size_t)n * 512 + 256 + (c - 32) * 8) = h;
  }
}

// ---------------------------------------------------------------------------
extern "C" void kernel_launch(void* const* d_in, const int* in_sizes, int n_in,
                              void* d_out, int out_size, void* d_ws, size_t ws_size,
                              hipStream_t stream) {
  (void)in_sizes; (void)n_in; (void)out_size; (void)ws_size;
  constexpr int N = GN_N, E = GN_E, L = GN_L, S = GN_S;

  const float* node_feats = (const float*)d_in[0];
  const float* edge_feats = (const float*)d_in[1];
  const int*   senders    = (const int*)d_in[2];
  const int*   receivers  = (const int*)d_in[3];
  const float* en_W0 = (const float*)d_in[4];
  const float* en_b0 = (const float*)d_in[5];
  const float* en_W1 = (const float*)d_in[6];
  const float* en_b1 = (const float*)d_in[7];
  const float* en_ls = (const float*)d_in[8];
  const float* en_lb = (const float*)d_in[9];
  const float* ee_W0 = (const float*)d_in[10];
  const float* ee_b0 = (const float*)d_in[11];
  const float* ee_W1 = (const float*)d_in[12];
  const float* ee_b1 = (const float*)d_in[13];
  const float* ee_ls = (const float*)d_in[14];
  const float* ee_lb = (const float*)d_in[15];
  const float* pe_W0 = (const float*)d_in[16];
  const float* pe_b0 = (const float*)d_in[17];
  const float* pe_W1 = (const float*)d_in[18];
  const float* pe_b1 = (const float*)d_in[19];
  const float* pe_ls = (const float*)d_in[20];
  const float* pe_lb = (const float*)d_in[21];
  const float* pn_W0 = (const float*)d_in[22];
  const float* pn_b0 = (const float*)d_in[23];
  const float* pn_W1 = (const float*)d_in[24];
  const float* pn_b1 = (const float*)d_in[25];
  const float* pn_ls = (const float*)d_in[26];
  const float* pn_lb = (const float*)d_in[27];
  const float* dn_W0 = (const float*)d_in[28];
  const float* dn_b0 = (const float*)d_in[29];
  const float* dn_W1 = (const float*)d_in[30];
  const float* dn_b1 = (const float*)d_in[31];

  // ---- workspace carving (~540 MB total) --------------------------------
  char* ws = (char*)d_ws;
  size_t off = 0;
  auto carve = [&](size_t bytes) -> void* {
    void* p = ws + off;
    off += (bytes + 255) & ~(size_t)255;
    return p;
  };
  auto pwsz = [](int Kp, int Nn) { return (size_t)(Nn / 16) * (Kp / 32) * 512 * 2; };

  _Float16* en_W0p = (_Float16*)carve(pwsz(32, 256));
  _Float16* en_W1p = (_Float16*)carve(pwsz(256, 256));
  _Float16* ee_W0p = (_Float16*)carve(pwsz(32, 256));
  _Float16* ee_W1p = (_Float16*)carve(pwsz(256, 256));
  _Float16* pe_W0p[S]; _Float16* pe_W1p[S]; _Float16* pn_W0p[S]; _Float16* pn_W1p[S];
  for (int s = 0; s < S; ++s) {
    pe_W0p[s] = (_Float16*)carve(pwsz(768, 256));
    pe_W1p[s] = (_Float16*)carve(pwsz(256, 256));
    pn_W0p[s] = (_Float16*)carve(pwsz(512, 256));
    pn_W1p[s] = (_Float16*)carve(pwsz(256, 256));
  }
  _Float16* dn_W0p = (_Float16*)carve(pwsz(256, 256));
  _Float16* dn_W1p = (_Float16*)carve(pwsz(256, 128));

  float*    nodes_f32 = (float*)carve((size_t)N * L * 4);
  _Float16* nodes_f16 = (_Float16*)carve((size_t)N * L * 2);
  float*    edges_f32 = (float*)carve((size_t)E * L * 4);
  _Float16* edges_f16 = (_Float16*)carve((size_t)E * L * 2);
  _Float16* ein       = (_Float16*)carve((size_t)E * 3 * L * 2); // 192MB
  float*    y_e       = (float*)ein;   // alias: ein dead once h_e is built
  _Float16* h_e       = (_Float16*)carve((size_t)E * L * 2);
  float*    agg       = (float*)carve((size_t)N * L * 4);
  _Float16* nin       = (_Float16*)carve((size_t)N * 2 * L * 2);
  _Float16* h_n       = (_Float16*)carve((size_t)N * L * 2);
  float*    y_n       = (float*)carve((size_t)N * L * 4);
  _Float16* a_nf      = (_Float16*)carve((size_t)N * 32 * 2);
  _Float16* a_ef      = (_Float16*)carve((size_t)E * 32 * 2);

  // ---- launch helpers ----------------------------------------------------
  auto pack = [&](const float* W, _Float16* dst, int K, int Kp, int Nn) {
    int total = (Nn / 16) * (Kp / 32) * 512;
    pack_weight_kernel<<<(total + 255) / 256, 256, 0, stream>>>(W, dst, K, Kp, Nn);
  };
  auto gemm_relu = [&](const _Float16* A, const _Float16* Bp, const float* b,
                       _Float16* C, int M, int Kp, int Nn) {
    dim3 g(M / 256, Nn / 64);
    gemm_wmma_kernel<0><<<g, 256, 0, stream>>>(A, Bp, b, (void*)C, M, Kp, Nn);
  };
  auto gemm_f32 = [&](const _Float16* A, const _Float16* Bp, const float* b,
                      float* C, int M, int Kp, int Nn) {
    dim3 g(M / 256, Nn / 64);
    gemm_wmma_kernel<1><<<g, 256, 0, stream>>>(A, Bp, b, (void*)C, M, Kp, Nn);
  };

  // ---- weight packing (L2-resident afterwards) ---------------------------
  pack(en_W0, en_W0p, 3, 32, 256);
  pack(en_W1, en_W1p, 256, 256, 256);
  pack(ee_W0, ee_W0p, 4, 32, 256);
  pack(ee_W1, ee_W1p, 256, 256, 256);
  for (int s = 0; s < S; ++s) {
    pack(pe_W0 + (size_t)s * 768 * 256, pe_W0p[s], 768, 768, 256);
    pack(pe_W1 + (size_t)s * 256 * 256, pe_W1p[s], 256, 256, 256);
    pack(pn_W0 + (size_t)s * 512 * 256, pn_W0p[s], 512, 512, 256);
    pack(pn_W1 + (size_t)s * 256 * 256, pn_W1p[s], 256, 256, 256);
  }
  pack(dn_W0, dn_W0p, 256, 256, 256);
  pack(dn_W1, dn_W1p, 256, 256, 128);

  // ---- encoder -----------------------------------------------------------
  convert_pad_kernel<<<(N * 32 + 255) / 256, 256, 0, stream>>>(node_feats, a_nf, 3, N);
  convert_pad_kernel<<<(E * 32 + 255) / 256, 256, 0, stream>>>(edge_feats, a_ef, 4, E);

  gemm_relu(a_nf, en_W0p, en_b0, h_n, N, 32, 256);
  gemm_f32(h_n, en_W1p, en_b1, y_n, N, 256, 256);
  layernorm_kernel<<<N / 8, 256, 0, stream>>>(y_n, en_ls, en_lb, nodes_f32, nodes_f16,
                                              nullptr, nullptr, N, 0);
  gemm_relu(a_ef, ee_W0p, ee_b0, h_e, E, 32, 256);
  gemm_f32(h_e, ee_W1p, ee_b1, y_e, E, 256, 256);
  layernorm_kernel<<<E / 8, 256, 0, stream>>>(y_e, ee_ls, ee_lb, edges_f32, edges_f16,
                                              nullptr, nullptr, E, 0);

  // ---- processor: S message-passing steps --------------------------------
  for (int s = 0; s < S; ++s) {
    zero_f4_kernel<<<(N * L / 4 + 255) / 256, 256, 0, stream>>>((float4*)agg, N * L / 4);
    gather_concat_kernel<<<(E * 96 + 255) / 256, 256, 0, stream>>>(
        edges_f16, nodes_f16, senders, receivers, ein, E);
    gemm_relu(ein, pe_W0p[s], pe_b0 + s * 256, h_e, E, 768, 256);
    gemm_f32(h_e, pe_W1p[s], pe_b1 + s * 256, y_e, E, 256, 256);  // overwrites ein (dead)
    // edge LN + residual + fused segment_sum(new_e -> agg[receivers])
    layernorm_kernel<<<E / 8, 256, 0, stream>>>(y_e, pe_ls + s * 256, pe_lb + s * 256,
                                                edges_f32, edges_f16, agg, receivers, E, 1);
    node_concat_kernel<<<(N * 64 + 255) / 256, 256, 0, stream>>>(nodes_f16, agg, nin, N);
    gemm_relu(nin, pn_W0p[s], pn_b0 + s * 256, h_n, N, 512, 256);
    gemm_f32(h_n, pn_W1p[s], pn_b1 + s * 256, y_n, N, 256, 256);
    layernorm_kernel<<<N / 8, 256, 0, stream>>>(y_n, pn_ls + s * 256, pn_lb + s * 256,
                                                nodes_f32, nodes_f16, nullptr, nullptr, N, 1);
  }

  // ---- decoder (no LN) ---------------------------------------------------
  gemm_relu(nodes_f16, dn_W0p, dn_b0, h_n, N, 256, 256);
  gemm_f32(h_n, dn_W1p, dn_b1, (float*)d_out, N, 256, 128);
}
